// CNF_29042568856021
// MI455X (gfx1250) — matrix operand order repaired and checked by reading, not compiled, not run on hardware
//
#include <hip/hip_runtime.h>

typedef __attribute__((ext_vector_type(16))) _Float16 v16h;
typedef __attribute__((ext_vector_type(8)))  float    v8f;

#define B_TOT 8192
#define D_DIM 64
#define H_DIM 256

// fragment region offsets (in _Float16 units) inside workspace
#define W1F_OFF 0          // 2 kt * 16 nt frags * 512 = 16384 halves (32 KB)
#define W2F_OFF 16384      // 8*16*512 = 65536 halves (128 KB)
#define EF_OFF  81920      // 8*16*512 = 65536 halves (128 KB)
#define W3F_OFF 147456     // 8*4*512  = 16384 halves (32 KB, streamed from L2)
#define FRAGS_TOTAL 163840
#define LDS_FRAG_HALVES 147456                 // W1F + W2F + EF staged in LDS
#define SMEM_BYTES (LDS_FRAG_HALVES*2 + (256+256+64)*4 + 8*256*4) // 305,408 B < 320 KB

#define LOG2E 1.4426950408889634f
#define LN2   0.6931471805599453f

static __device__ __forceinline__ v8f wmma_f16(v16h a, v16h b, v8f c) {
  // D = A(16x32 f16) * B(32x16 f16) + C(16x16 f32)
  return __builtin_amdgcn_wmma_f32_16x16x32_f16(false, a, false, b, (short)0, c,
                                                false, false);
}

static __device__ __forceinline__ void put4(v16h& f, int p, float4 q) {
  f[p+0] = (_Float16)q.x; f[p+1] = (_Float16)q.y;
  f[p+2] = (_Float16)q.z; f[p+3] = (_Float16)q.w;
}

// softplus via raw v_exp_f32 / v_log_f32 (base-2): sp = max(z,0) + ln2*log2(1+2^(-|z|*log2e))
static __device__ __forceinline__ float fast_softplus(float z) {
  float e2 = __builtin_amdgcn_exp2f(-fabsf(z) * LOG2E);
  return fmaxf(z, 0.f) + LN2 * __builtin_amdgcn_logf(1.f + e2);
}
// fused softplus + sigmoid reusing the same exp2 term:
//   r = 1/(1+e2);  sigma = (z>=0) ? r : 1-r
static __device__ __forceinline__ float fast_softplus_sig(float z, float& sig) {
  float e2 = __builtin_amdgcn_exp2f(-fabsf(z) * LOG2E);
  float r  = __builtin_amdgcn_rcpf(1.f + e2);
  sig = (z >= 0.f) ? r : 1.f - r;
  return fmaxf(z, 0.f) + LN2 * __builtin_amdgcn_logf(1.f + e2);
}

// ---------------------------------------------------------------------------
// Precompute: swizzle W1^T, W2^T, E^T, W3^T into WMMA-B fragment layout (f16).
//   B-frag layout (32x16 f16, 8 VGPRs): lane L holds column n = L%16,
//   half h holds K = (L&16)/2 + (h&7) + 2*(h&8) within the 32-wide K slab.
//   E[n,k] = W2[n,k] * M[k,n],  M = W1 @ W3  (closed-form Jacobian trace).
// ---------------------------------------------------------------------------
__global__ void prep_frags(const float* __restrict__ W1, const float* __restrict__ W2,
                           const float* __restrict__ W3, _Float16* __restrict__ frags) {
  int tid = blockIdx.x * blockDim.x + threadIdx.x;
  if (tid >= FRAGS_TOTAL) return;
  float val = 0.f;
  if (tid < W2F_OFF) {                       // W1F: B[k,n] = W1[n,k], k in D, n in H
    int idx = tid, f = idx >> 9, rem = idx & 511, lane = rem >> 4, h = rem & 15;
    int kt = f >> 4, nt = f & 15;
    int k = kt * 32 + ((lane & 16) >> 1) + ((h & 8) << 1) + (h & 7);
    int n = nt * 16 + (lane & 15);
    val = W1[n * D_DIM + k];
  } else if (tid < EF_OFF) {                 // W2F: B[k,n] = W2[n,k]
    int idx = tid - W2F_OFF, f = idx >> 9, rem = idx & 511, lane = rem >> 4, h = rem & 15;
    int kt = f >> 4, nt = f & 15;
    int k = kt * 32 + ((lane & 16) >> 1) + ((h & 8) << 1) + (h & 7);
    int n = nt * 16 + (lane & 15);
    val = W2[n * H_DIM + k];
  } else if (tid < W3F_OFF) {                // EF: B[k,n] = W2[n,k]*M[k,n]
    int idx = tid - EF_OFF, f = idx >> 9, rem = idx & 511, lane = rem >> 4, h = rem & 15;
    int kt = f >> 4, nt = f & 15;
    int k = kt * 32 + ((lane & 16) >> 1) + ((h & 8) << 1) + (h & 7);
    int n = nt * 16 + (lane & 15);
    float m = 0.f;
    for (int d = 0; d < D_DIM; ++d) m += W1[k * D_DIM + d] * W3[d * H_DIM + n];
    val = W2[n * H_DIM + k] * m;
  } else {                                   // W3F: B[k,n] = W3[n,k], k in H, n in D
    int idx = tid - W3F_OFF, f = idx >> 9, rem = idx & 511, lane = rem >> 4, h = rem & 15;
    int kt = f >> 2, nt = f & 3;
    int k = kt * 32 + ((lane & 16) >> 1) + ((h & 8) << 1) + (h & 7);
    int n = nt * 16 + (lane & 15);
    val = W3[n * H_DIM + k];
  }
  frags[tid] = (_Float16)val;
}

// ---------------------------------------------------------------------------
// dyn_eval: one wave = one 16-row batch tile. 320 v_wmma per wave.
//   GEMM1: z1 = y @ W1^T + b1         (16x64  * 64x256)
//   GEMM2: z2 = sp(z1) @ W2^T + b2    (16x256 * 256x256)
//   GEMM4: t  = sig(z1) @ E^T         (16x256 * 256x256)   trace = sum sig(z2)*t
//   GEMM3: dy = sp(z2) @ W3^T + b3    (16x256 * 256x64)
// sigma1 derived once from softplus1 fragments: sig = 1 - 2^(-sp*log2e).
// ---------------------------------------------------------------------------
__global__ void __launch_bounds__(256)
dyn_eval(const float* __restrict__ ys, const _Float16* __restrict__ frags,
         const float* __restrict__ b1, const float* __restrict__ b2,
         const float* __restrict__ b3, float* __restrict__ ky,
         float* __restrict__ kl) {
  extern __shared__ char smem_raw[];
  _Float16* sfrag = (_Float16*)smem_raw;                       // 294,912 B
  float* sb1 = (float*)(smem_raw + LDS_FRAG_HALVES * 2);
  float* sb2 = sb1 + 256;
  float* sb3 = sb2 + 256;
  float* scr = (float*)(smem_raw + LDS_FRAG_HALVES * 2 + 2304); // 8 waves * 16x16 f32

  // stage weight fragments into LDS (cooperative, 16B chunks)
  {
    const float4* g = (const float4*)frags;
    float4* l = (float4*)sfrag;
    for (int i = threadIdx.x; i < (LDS_FRAG_HALVES * 2) / 16; i += 256) l[i] = g[i];
    sb1[threadIdx.x] = b1[threadIdx.x];
    sb2[threadIdx.x] = b2[threadIdx.x];
    if (threadIdx.x < 64) sb3[threadIdx.x] = b3[threadIdx.x];
  }
  __syncthreads();

  const int lane = threadIdx.x & 31;
  const int wave = threadIdx.x >> 5;
  const int row0 = (blockIdx.x * 8 + wave) * 16;
  const int mrow = lane & 15;    // A-frag row / D-tile column
  const int kgrp = lane >> 4;    // K-slab selector (A) / row-half (C/D)
  float* myscr = scr + wave * 256;

  // ---- y tile -> A fragments (f16). A layout: lane holds row mrow,
  //      halves 0..7 -> K = kgrp*8 + 0..7, halves 8..15 -> K = 16+kgrp*8 + 0..7
  v16h yf[2];
  {
    const float* src = ys + (size_t)(row0 + mrow) * D_DIM;
#pragma unroll
    for (int kt = 0; kt < 2; ++kt) {
      const float* s0 = src + kt * 32 + kgrp * 8;
      const float* s1 = src + kt * 32 + 16 + kgrp * 8;
      v16h f;
      put4(f, 0,  *(const float4*)(s0));
      put4(f, 4,  *(const float4*)(s0 + 4));
      put4(f, 8,  *(const float4*)(s1));
      put4(f, 12, *(const float4*)(s1 + 4));
      yf[kt] = f;
    }
  }

  // ---- GEMM1 + softplus; D-layout -> A-layout via per-wave LDS scratch ----
  v16h h1f[8];
#pragma unroll
  for (int nt = 0; nt < 16; ++nt) {
    v8f c = {0.f, 0.f, 0.f, 0.f, 0.f, 0.f, 0.f, 0.f};
    {
      v16h bw0 = *(const v16h*)(sfrag + W1F_OFF + (size_t)((0 * 16 + nt) * 32 + lane) * 16);
      v16h bw1 = *(const v16h*)(sfrag + W1F_OFF + (size_t)((1 * 16 + nt) * 32 + lane) * 16);
      c = wmma_f16(yf[0], bw0, c);
      c = wmma_f16(yf[1], bw1, c);
    }
    float bias = sb1[nt * 16 + mrow];
#pragma unroll
    for (int r = 0; r < 8; ++r) {            // D elem r -> row kgrp*8+r, col mrow
      float z = c[r] + bias;
      myscr[(kgrp * 8 + r) * 16 + mrow] = fast_softplus(z);
    }
    asm volatile("s_wait_dscnt 0" ::: "memory");
    float4 g0 = *(const float4*)(myscr + mrow * 16 + kgrp * 8);
    float4 g1 = *(const float4*)(myscr + mrow * 16 + kgrp * 8 + 4);
    const int fi = nt >> 1, hb = (nt & 1) * 8;
    put4(h1f[fi], hb, g0);
    put4(h1f[fi], hb + 4, g1);
  }

  // ---- sigma1 fragments, computed ONCE: sig1 = 1 - 2^(-sp1*log2e) ----
  v16h s1f[8];
#pragma unroll
  for (int kt = 0; kt < 8; ++kt) {
#pragma unroll
    for (int i = 0; i < 16; ++i) {
      float h = (float)h1f[kt][i];
      s1f[kt][i] = (_Float16)(1.f - __builtin_amdgcn_exp2f(-h * LOG2E));
    }
  }

  // ---- fused GEMM2 (W2) + GEMM4 (E) + trace accumulation ----
  v16h h2f[8];
  float tr[8];
#pragma unroll
  for (int r = 0; r < 8; ++r) tr[r] = 0.f;
#pragma unroll
  for (int nt = 0; nt < 16; ++nt) {
    v8f c2 = {0.f, 0.f, 0.f, 0.f, 0.f, 0.f, 0.f, 0.f};
    v8f c4 = {0.f, 0.f, 0.f, 0.f, 0.f, 0.f, 0.f, 0.f};
    // double-buffered B-fragment loads: fetch kt+1 while multiplying kt
    v16h bw = *(const v16h*)(sfrag + W2F_OFF + (size_t)((0 * 16 + nt) * 32 + lane) * 16);
    v16h be = *(const v16h*)(sfrag + EF_OFF  + (size_t)((0 * 16 + nt) * 32 + lane) * 16);
#pragma unroll
    for (int kt = 0; kt < 8; ++kt) {
      v16h bwn, ben;
      if (kt < 7) {
        bwn = *(const v16h*)(sfrag + W2F_OFF + (size_t)(((kt + 1) * 16 + nt) * 32 + lane) * 16);
        ben = *(const v16h*)(sfrag + EF_OFF  + (size_t)(((kt + 1) * 16 + nt) * 32 + lane) * 16);
      }
      c2 = wmma_f16(h1f[kt], bw, c2);
      c4 = wmma_f16(s1f[kt], be, c4);
      bw = bwn; be = ben;
    }
    float bias = sb2[nt * 16 + mrow];
#pragma unroll
    for (int r = 0; r < 8; ++r) {
      float z = c2[r] + bias;
      float sg;
      float sp = fast_softplus_sig(z, sg);
      tr[r] += sg * c4[r];                   // sigma2 .* (E @ sigma1)
      myscr[(kgrp * 8 + r) * 16 + mrow] = sp;
    }
    asm volatile("s_wait_dscnt 0" ::: "memory");
    float4 g0 = *(const float4*)(myscr + mrow * 16 + kgrp * 8);
    float4 g1 = *(const float4*)(myscr + mrow * 16 + kgrp * 8 + 4);
    const int fi = nt >> 1, hb = (nt & 1) * 8;
    put4(h2f[fi], hb, g0);
    put4(h2f[fi], hb + 4, g1);
  }

  // ---- GEMM3: dy = h2 @ W3^T + b3 (W3 frags streamed from global/L2) ----
#pragma unroll
  for (int nt = 0; nt < 4; ++nt) {
    v8f c = {0.f, 0.f, 0.f, 0.f, 0.f, 0.f, 0.f, 0.f};
    v16h bw = *(const v16h*)(frags + W3F_OFF + (size_t)((0 * 4 + nt) * 32 + lane) * 16);
#pragma unroll
    for (int kt = 0; kt < 8; ++kt) {
      v16h bwn;
      if (kt < 7)
        bwn = *(const v16h*)(frags + W3F_OFF + (size_t)(((kt + 1) * 4 + nt) * 32 + lane) * 16);
      c = wmma_f16(h2f[kt], bw, c);
      bw = bwn;
    }
    float bias = sb3[nt * 16 + mrow];
#pragma unroll
    for (int r = 0; r < 8; ++r)
      ky[(size_t)(row0 + kgrp * 8 + r) * D_DIM + nt * 16 + mrow] = c[r] + bias;
  }

  // ---- trace: reduce over the 16 lanes of each row-half group ----
#pragma unroll
  for (int r = 0; r < 8; ++r) {
    float v = tr[r];
    v += __shfl_xor(v, 1, 32);
    v += __shfl_xor(v, 2, 32);
    v += __shfl_xor(v, 4, 32);
    v += __shfl_xor(v, 8, 32);
    tr[r] = v;
  }
  if (mrow == 0) {
#pragma unroll
    for (int r = 0; r < 8; ++r) kl[row0 + kgrp * 8 + r] = tr[r];
  }
}

// ---------------------------------------------------------------------------
// Butcher-tableau combine: out = base + (T/NUM_STEPS) * sum_j cj * kj
// ---------------------------------------------------------------------------
__global__ void combine(const float* __restrict__ base, const float* __restrict__ Tptr,
                        const float* __restrict__ k1, const float* __restrict__ k2,
                        const float* __restrict__ k3, const float* __restrict__ k4,
                        const float* __restrict__ k5, const float* __restrict__ k6,
                        float c1, float c2, float c3, float c4, float c5, float c6,
                        float* __restrict__ out, int n) {
  int i = blockIdx.x * blockDim.x + threadIdx.x;
  if (i >= n) return;
  float dt = Tptr[0] * 0.1f;
  float acc = c1 * k1[i] + c2 * k2[i] + c3 * k3[i] +
              c4 * k4[i] + c5 * k5[i] + c6 * k6[i];
  out[i] = base[i] + dt * acc;
}

extern "C" void kernel_launch(void* const* d_in, const int* in_sizes, int n_in,
                              void* d_out, int out_size, void* d_ws, size_t ws_size,
                              hipStream_t stream) {
  const float* x   = (const float*)d_in[0];
  const float* ld0 = (const float*)d_in[1];
  const float* W1  = (const float*)d_in[2];
  const float* b1  = (const float*)d_in[3];
  const float* W2  = (const float*)d_in[4];
  const float* b2  = (const float*)d_in[5];
  const float* W3  = (const float*)d_in[6];
  const float* b3  = (const float*)d_in[7];
  const float* T   = (const float*)d_in[8];

  float* y  = (float*)d_out;                         // running y  [B, 64]
  float* ld = (float*)d_out + (size_t)B_TOT * D_DIM; // running logdet [B]

  char* ws = (char*)d_ws;
  _Float16* frags = (_Float16*)ws;
  const size_t nY = (size_t)B_TOT * D_DIM;
  float* ysb = (float*)(ws + (size_t)FRAGS_TOTAL * 2);
  float* p = ysb + nY;
  float* kY[6]; float* kL[6];
  for (int i = 0; i < 6; ++i) { kY[i] = p; p += nY; }
  for (int i = 0; i < 6; ++i) { kL[i] = p; p += B_TOT; }

  prep_frags<<<FRAGS_TOTAL / 256, 256, 0, stream>>>(W1, W2, W3, frags);

  auto CMB = [&](const float* base, float* out, size_t n,
                 const float* p1, const float* p2, const float* p3,
                 const float* p4, const float* p5, const float* p6,
                 float q1, float q2, float q3, float q4, float q5, float q6) {
    combine<<<(int)((n + 255) / 256), 256, 0, stream>>>(
        base, T, p1, p2, p3, p4, p5, p6, q1, q2, q3, q4, q5, q6, out, (int)n);
  };

  // init running state: y = x, ld = logdet_init
  CMB(x, y, nY, x, x, x, x, x, x, 0, 0, 0, 0, 0, 0);
  CMB(ld0, ld, B_TOT, ld0, ld0, ld0, ld0, ld0, ld0, 0, 0, 0, 0, 0, 0);

  auto DYN = [&](const float* yin, int s) {
    dyn_eval<<<B_TOT / 128, 256, SMEM_BYTES, stream>>>(yin, frags, b1, b2, b3,
                                                       kY[s], kL[s]);
  };

  const float a21 = 0.2f;
  const float a31 = 3.f / 40.f, a32 = 9.f / 40.f;
  const float a41 = 44.f / 45.f, a42 = -56.f / 15.f, a43 = 32.f / 9.f;
  const float a51 = 19372.f / 6561.f, a52 = -25360.f / 2187.f,
              a53 = 64448.f / 6561.f, a54 = -212.f / 729.f;
  const float a61 = 9017.f / 3168.f, a62 = -355.f / 33.f,
              a63 = 46732.f / 5247.f, a64 = 49.f / 176.f, a65 = -5103.f / 18656.f;
  const float c1 = 35.f / 384.f, c3 = 500.f / 1113.f, c4 = 125.f / 192.f,
              c5 = -2187.f / 6784.f, c6 = 11.f / 84.f;

  for (int step = 0; step < 10; ++step) {
    DYN(y, 0);
    CMB(y, ysb, nY, kY[0], kY[0], kY[0], kY[0], kY[0], kY[0], a21, 0, 0, 0, 0, 0);
    DYN(ysb, 1);
    CMB(y, ysb, nY, kY[0], kY[1], kY[0], kY[0], kY[0], kY[0], a31, a32, 0, 0, 0, 0);
    DYN(ysb, 2);
    CMB(y, ysb, nY, kY[0], kY[1], kY[2], kY[0], kY[0], kY[0], a41, a42, a43, 0, 0, 0);
    DYN(ysb, 3);
    CMB(y, ysb, nY, kY[0], kY[1], kY[2], kY[3], kY[0], kY[0], a51, a52, a53, a54, 0, 0);
    DYN(ysb, 4);
    CMB(y, ysb, nY, kY[0], kY[1], kY[2], kY[3], kY[4], kY[0], a61, a62, a63, a64, a65, 0);
    DYN(ysb, 5);
    // 5th-order update (in place)
    CMB(y, y, nY, kY[0], kY[1], kY[2], kY[3], kY[4], kY[5], c1, 0, c3, c4, c5, c6);
    CMB(ld, ld, B_TOT, kL[0], kL[1], kL[2], kL[3], kL[4], kL[5], c1, 0, c3, c4, c5, c6);
  }
}
